// H2OLlamaAttention_55714315764324
// MI455X (gfx1250) — matrix operand order repaired and checked by reading, not compiled, not run on hardware
//
#include <hip/hip_runtime.h>
#include <hip/hip_bf16.h>
#include <stdint.h>

// Problem constants (match reference)
#define S_LEN 4096
#define HID   2048
#define NH    16
#define NKV   4
#define HD    128
#define KVDIM (NKV * HD)   // 512

typedef __attribute__((ext_vector_type(16))) __bf16 v16bf;
typedef __attribute__((ext_vector_type(8)))  float  v8f;
typedef __attribute__((ext_vector_type(4)))  unsigned int v4u;
typedef __attribute__((ext_vector_type(8)))  int v8i;
typedef __attribute__((ext_vector_type(4)))  int v4i;

union FragCast { struct { uint4 lo, hi; } u; v16bf v; };

static __device__ __forceinline__ v16bf make_frag(uint4 a, uint4 b) {
  FragCast f; f.u.lo = a; f.u.hi = b; return f.v;
}

// f32 -> bf16 round-to-nearest-even via bit ops (storage is uint16)
static __device__ __forceinline__ unsigned short f2bf(float x) {
  unsigned u = __float_as_uint(x);
  u += 0x7FFFu + ((u >> 16) & 1u);
  return (unsigned short)(u >> 16);
}
static __device__ __forceinline__ float bf2f(unsigned short h) {
  return __uint_as_float(((unsigned)h) << 16);
}

// ---------------------------------------------------------------------------
// Tensor Data Mover: 2D tile load Global -> LDS (CDNA5 ISA ch.8 D# layout).
// data_size = 2 bytes (bf16). Descriptor groups built in scalar regs:
//   group0: count=1 | lds_addr | global_addr(57b) | type=2
//   group1: data_size=1(2B), pad_enable, pad_interval/amount, dims, strides
// Groups 2/3 (+extra group on this toolchain) zero: 2D tile only.
// Issued by one wave; completion via TENSORcnt (s_wait_tensorcnt).
// ---------------------------------------------------------------------------
static __device__ __forceinline__ void tdm_load_2d(
    const void* gptr, unsigned lds_addr,
    unsigned tensor_d0, unsigned tensor_d1,
    unsigned tile_d0, unsigned tile_d1,
    unsigned long long stride0_elems,
    unsigned pad_interval_code, unsigned pad_amount_code) {
  unsigned long long ga = (unsigned long long)(uintptr_t)gptr;
  v4u g0;
  g0[0] = 1u;                                   // count=1, user descriptor
  g0[1] = lds_addr;                             // LDS byte address
  g0[2] = (unsigned)ga;                         // global addr [31:0]
  g0[3] = (unsigned)((ga >> 32) & 0x01FFFFFFu)  // global addr [56:32]
          | (2u << 30);                         // type = 2 ("image")
  v8i g1;
  g1[0] = (int)((1u << 16)                      // data_size = 1 -> 2 bytes
                | (1u << 20)                    // pad_enable
                | (pad_interval_code << 22)
                | (pad_amount_code << 25));
  g1[1] = (int)((tensor_d0 & 0xFFFFu) << 16);                       // dim0 lo
  g1[2] = (int)((tensor_d0 >> 16) | ((tensor_d1 & 0xFFFFu) << 16)); // dim0 hi|dim1 lo
  g1[3] = (int)((tensor_d1 >> 16) | ((tile_d0 & 0xFFFFu) << 16));   // dim1 hi|tile0
  g1[4] = (int)(tile_d1 & 0xFFFFu);                                 // tile1, tile2=0
  g1[5] = (int)(stride0_elems & 0xFFFFFFFFull);                     // dim0 stride lo
  g1[6] = (int)((stride0_elems >> 32) & 0xFFFFull);                 // stride hi|t1s lo
  g1[7] = 0;                                                        // t1s hi (unused 2D)
  v4i z4 = {0, 0, 0, 0};
  v8i z8 = {0, 0, 0, 0, 0, 0, 0, 0};
  __builtin_amdgcn_tensor_load_to_lds(g0, g1, z4, z4, z8, 0);
}

// ---------------------------------------------------------------------------
// Elementwise f32 -> bf16 conversion, 8 elements/thread, 16B stores
// ---------------------------------------------------------------------------
__global__ void cvt_f32_bf16(const float* __restrict__ src,
                             unsigned short* __restrict__ dst, int n8) {
  int i = blockIdx.x * blockDim.x + threadIdx.x;
  if (i >= n8) return;
  const float4* s = (const float4*)src + 2 * (size_t)i;
  float4 a = s[0], b = s[1];
  uint4 r;
  r.x = (unsigned)f2bf(a.x) | ((unsigned)f2bf(a.y) << 16);
  r.y = (unsigned)f2bf(a.z) | ((unsigned)f2bf(a.w) << 16);
  r.z = (unsigned)f2bf(b.x) | ((unsigned)f2bf(b.y) << 16);
  r.w = (unsigned)f2bf(b.z) | ((unsigned)f2bf(b.w) << 16);
  *(uint4*)(dst + 8 * (size_t)i) = r;
}

// f32 [K][N] -> bf16 transposed [N][K] (one-time weight layout fix so GEMM
// B-fragments become contiguous reads matching the WMMA B register layout)
__global__ void cvt_f32_bf16_T(const float* __restrict__ src,
                               unsigned short* __restrict__ dst,
                               int K, int N) {
  int i = blockIdx.x * blockDim.x + threadIdx.x;
  int row8 = N >> 3;
  int k = i / row8;
  if (k >= K) return;
  int n0 = (i - k * row8) << 3;
  const float4* s = (const float4*)(src + (size_t)k * N + n0);
  float4 a = s[0], b = s[1];
  float v[8] = {a.x, a.y, a.z, a.w, b.x, b.y, b.z, b.w};
#pragma unroll
  for (int j = 0; j < 8; ++j)
    dst[(size_t)(n0 + j) * K + k] = f2bf(v[j]);
}

// ---------------------------------------------------------------------------
// bf16 GEMM: C[M x N] = A[M x K] * Bt^T, A row-major, Bt PRE-TRANSPOSED [N][K].
// 256 threads = 8 waves; block tile 128x128, K step 32.
// A tile staged by the Tensor Data Mover into padded LDS (stride 40 halves,
// conflict-free b128 fragment reads), double-buffered against the 8 WMMAs.
// B fragments read directly from global (weights are L2-resident, 32x reuse).
// OUTMODE: 0 = bf16, 1 = f32, 2 = bf16 transposed (C[col*S_LEN + row])
// ---------------------------------------------------------------------------
template <int OUTMODE>
__global__ __launch_bounds__(256)
void gemm_bf16(const unsigned short* __restrict__ A,
               const unsigned short* __restrict__ Bt,
               void* __restrict__ C, int N, int K) {
  __shared__ unsigned short As[2][128 * 40];  // [row][k], TDM-padded stride 40

  const int tid  = threadIdx.x;
  const int lane = tid & 31;
  const int wave = tid >> 5;
  const int wm   = wave & 3;   // 0..3 -> 32-row slot
  const int wn   = wave >> 2;  // 0..1 -> 64-col slot
  const int hl   = lane & 15;
  const int m0   = blockIdx.y * 128;
  const int n0   = blockIdx.x * 128;

  v8f acc[2][4] = {};

  // prologue: DMA first A tile (128 rows x 32 k), pad 16dw-interval by 4dw
  if (wave == 0) {
    tdm_load_2d(A + (size_t)m0 * K, (unsigned)(uintptr_t)&As[0][0],
                (unsigned)K, S_LEN, 32, 128, (unsigned long long)K,
                /*interval 16dw*/ 3, /*amount 4dw*/ 3);
    __builtin_amdgcn_s_wait_tensorcnt(0);
  }
  __syncthreads();

  const int akb = (lane < 16) ? 0 : 8;   // A frag: lane<16 K{0..7,16..23}
  const int bkb = (lane < 16) ? 0 : 8;   // B frag: lane<16 K{0..7,16..23}

  for (int k0 = 0; k0 < K; k0 += 32) {
    const int cur = (k0 >> 5) & 1;
    if (wave == 0 && k0 + 32 < K)  // overlap DMA of next tile with compute
      tdm_load_2d(A + (size_t)m0 * K + (k0 + 32),
                  (unsigned)(uintptr_t)&As[cur ^ 1][0],
                  (unsigned)K, S_LEN, 32, 128, (unsigned long long)K, 3, 3);

    v16bf af[2], bfr[4];
#pragma unroll
    for (int mf = 0; mf < 2; ++mf) {
      int row = wm * 32 + mf * 16 + hl;
      af[mf] = make_frag(*(const uint4*)&As[cur][row * 40 + akb],
                         *(const uint4*)&As[cur][row * 40 + akb + 16]);
    }
#pragma unroll
    for (int nf = 0; nf < 4; ++nf) {
      const unsigned short* bp =
          Bt + (size_t)(n0 + wn * 64 + nf * 16 + hl) * K + k0 + bkb;
      bfr[nf] = make_frag(*(const uint4*)bp, *(const uint4*)(bp + 16));
      if (k0 + 32 < K) __builtin_prefetch(bp + 32, 0, 0);
    }
#pragma unroll
    for (int mf = 0; mf < 2; ++mf)
#pragma unroll
      for (int nf = 0; nf < 4; ++nf)
        acc[mf][nf] = __builtin_amdgcn_wmma_f32_16x16x32_bf16(
            false, af[mf], false, bfr[nf], (short)0, acc[mf][nf], false, false);

    if (wave == 0) __builtin_amdgcn_s_wait_tensorcnt(0);
    __syncthreads();
  }

  // epilogue: C layout lane%16 = N col; vgpr r = row (lane>=16 -> +8)
  const int rofs = (lane >> 4) * 8;
#pragma unroll
  for (int mf = 0; mf < 2; ++mf)
#pragma unroll
    for (int nf = 0; nf < 4; ++nf) {
      int col = n0 + wn * 64 + nf * 16 + hl;
#pragma unroll
      for (int r = 0; r < 8; ++r) {
        int row = m0 + wm * 32 + mf * 16 + rofs + r;
        float v = acc[mf][nf][r];
        if (OUTMODE == 1)
          ((float*)C)[(size_t)row * N + col] = v;
        else if (OUTMODE == 2)  // transposed bf16 (V^T for flash attention)
          ((unsigned short*)C)[(size_t)col * S_LEN + row] = f2bf(v);
        else
          ((unsigned short*)C)[(size_t)row * N + col] = f2bf(v);
      }
    }
}

// ---------------------------------------------------------------------------
// RoPE in-place on bf16 [S][nheads*128]; one thread per (s, head, d<64) pair
// ---------------------------------------------------------------------------
__global__ void rope_kernel(unsigned short* __restrict__ X,
                            const int* __restrict__ pos_ids, int nheads) {
  int idx = blockIdx.x * blockDim.x + threadIdx.x;
  int per_s = nheads * 64;
  int s = idx / per_s;
  if (s >= S_LEN) return;
  int rem = idx - s * per_s;
  int h = rem >> 6;
  int d = rem & 63;
  float pos = (float)pos_ids[s];
  float inv = __expf(-(float)(2 * d) * (9.210340371976184f / 128.0f));
  float sn, cs;
  __sincosf(pos * inv, &sn, &cs);
  size_t base = (size_t)s * (nheads * HD) + (size_t)h * HD + d;
  float x1 = bf2f(X[base]);
  float x2 = bf2f(X[base + 64]);
  X[base]      = f2bf(x1 * cs - x2 * sn);
  X[base + 64] = f2bf(x2 * cs + x1 * sn);
}

// ---------------------------------------------------------------------------
// Causal flash attention. Grid (S/64, NH), block 128 = 4 waves.
// Each wave owns 16 Q rows; KV in 32-column steps.
// K tile TDM-staged to LDS (row-major IS the QK^T B layout), double-buffered.
// V read directly from the pre-transposed V^T [KVDIM][S] buffer.
// ---------------------------------------------------------------------------
__global__ __launch_bounds__(128)
void flash_attn(const unsigned short* __restrict__ Q,
                const unsigned short* __restrict__ Km,
                const unsigned short* __restrict__ Vt,   // [KVDIM][S_LEN]
                unsigned short* __restrict__ O) {
  __shared__ unsigned short Ks[2][32 * 136];  // K tile, TDM-padded stride 136
  __shared__ unsigned short Pb[4 * 16 * 40];  // per-wave 16x32 P tile

  const int tid  = threadIdx.x;
  const int lane = tid & 31;
  const int wave = tid >> 5;
  const int hl   = lane & 15;
  const int h    = blockIdx.y;
  const int kvh  = h / (NH / NKV);
  const int qbase = blockIdx.x * 64 + wave * 16;
  const int qmax  = blockIdx.x * 64 + 63;
  const int rofs  = (lane >> 4) * 8;
  const int pb    = wave * 16 * 40;
  const float scale = 0.08838834764831845f;  // 1/sqrt(128)

  // Q fragments held in registers for the whole KV loop (16 x 128 = 4 A-frags)
  v16bf qf[4];
  {
    const unsigned short* qp = Q + (size_t)(qbase + hl) * HID + (size_t)h * HD;
    int kb = (lane < 16) ? 0 : 8;
#pragma unroll
    for (int c = 0; c < 4; ++c)
      qf[c] = make_frag(*(const uint4*)(qp + c * 32 + kb),
                        *(const uint4*)(qp + c * 32 + kb + 16));
  }

  v8f of[8] = {};
  float mrun[8], lrun[8];
#pragma unroll
  for (int r = 0; r < 8; ++r) { mrun[r] = -1e30f; lrun[r] = 0.0f; }

  // prologue: DMA first K tile (32 kv rows x 128 dims), 64dw rows pad by 4dw
  if (wave == 0) {
    tdm_load_2d(Km + (size_t)kvh * HD, (unsigned)(uintptr_t)&Ks[0][0],
                KVDIM, S_LEN, 128, 32, (unsigned long long)KVDIM,
                /*interval 64dw*/ 5, /*amount 4dw*/ 3);
    __builtin_amdgcn_s_wait_tensorcnt(0);
  }
  __syncthreads();

  const int kkb = (lane < 16) ? 0 : 8;
  const int vkb = (lane < 16) ? 0 : 8;
  const int pkb = (lane < 16) ? 0 : 8;

  for (int kb0 = 0; kb0 <= qmax; kb0 += 32) {
    const int cur = (kb0 >> 5) & 1;
    if (wave == 0 && kb0 + 32 <= qmax)  // overlap next K tile DMA with compute
      tdm_load_2d(Km + (size_t)(kb0 + 32) * KVDIM + (size_t)kvh * HD,
                  (unsigned)(uintptr_t)&Ks[cur ^ 1][0],
                  KVDIM, S_LEN, 128, 32, (unsigned long long)KVDIM, 5, 3);

    // S = Q * K^T for two 16-wide kv groups (LDS rows are the B layout)
    v8f sacc[2] = {};
#pragma unroll
    for (int g = 0; g < 2; ++g) {
      const unsigned short* kp = &Ks[cur][(g * 16 + hl) * 136];
#pragma unroll
      for (int c = 0; c < 4; ++c) {
        v16bf kf = make_frag(*(const uint4*)(kp + c * 32 + kkb),
                             *(const uint4*)(kp + c * 32 + kkb + 16));
        sacc[g] = __builtin_amdgcn_wmma_f32_16x16x32_bf16(
            false, qf[c], false, kf, (short)0, sacc[g], false, false);
      }
    }

    // scale + causal mask + online softmax; stash P (bf16) in per-wave LDS
#pragma unroll
    for (int r = 0; r < 8; ++r) {
      int row = qbase + rofs + r;
      float s0 = sacc[0][r] * scale;
      float s1 = sacc[1][r] * scale;
      if (kb0 + hl > row)      s0 = -1e30f;
      if (kb0 + 16 + hl > row) s1 = -1e30f;
      float mx = fmaxf(s0, s1);
#pragma unroll
      for (int msk = 1; msk < 16; msk <<= 1)
        mx = fmaxf(mx, __shfl_xor(mx, msk, 32));
      float nm   = fmaxf(mrun[r], mx);
      float corr = __expf(mrun[r] - nm);
      float e0 = __expf(s0 - nm), e1 = __expf(s1 - nm);
      float rs = e0 + e1;
#pragma unroll
      for (int msk = 1; msk < 16; msk <<= 1) rs += __shfl_xor(rs, msk, 32);
      lrun[r] = lrun[r] * corr + rs;
      mrun[r] = nm;
#pragma unroll
      for (int dcf = 0; dcf < 8; ++dcf) of[dcf][r] *= corr;
      Pb[pb + (rofs + r) * 40 + hl]      = f2bf(e0);
      Pb[pb + (rofs + r) * 40 + 16 + hl] = f2bf(e1);
    }

    // reload P as a 16x32 A fragment (same-wave LDS RAW, ds ops in-order)
    v16bf pf = make_frag(*(const uint4*)&Pb[pb + hl * 40 + pkb],
                         *(const uint4*)&Pb[pb + hl * 40 + pkb + 16]);

    // O += P * V : V^T rows are contiguous -> direct global B fragments
#pragma unroll
    for (int dcf = 0; dcf < 8; ++dcf) {
      const unsigned short* vp =
          Vt + (size_t)(kvh * HD + dcf * 16 + hl) * S_LEN + kb0 + vkb;
      v16bf vf = make_frag(*(const uint4*)vp, *(const uint4*)(vp + 16));
      of[dcf] = __builtin_amdgcn_wmma_f32_16x16x32_bf16(
          false, pf, false, vf, (short)0, of[dcf], false, false);
    }

    if (wave == 0) __builtin_amdgcn_s_wait_tensorcnt(0);
    __syncthreads();  // release Ks buffer for next DMA
  }

  // epilogue: normalize by l and write bf16 [S][NH*HD]
#pragma unroll
  for (int r = 0; r < 8; ++r) {
    float invl = 1.0f / lrun[r];
    unsigned short* op = O + (size_t)(qbase + rofs + r) * HID + (size_t)h * HD;
#pragma unroll
    for (int dcf = 0; dcf < 8; ++dcf)
      op[dcf * 16 + hl] = f2bf(of[dcf][r] * invl);
  }
}

// ---------------------------------------------------------------------------
// Host launcher
// ---------------------------------------------------------------------------
extern "C" void kernel_launch(void* const* d_in, const int* in_sizes, int n_in,
                              void* d_out, int out_size, void* d_ws,
                              size_t ws_size, hipStream_t stream) {
  (void)in_sizes; (void)n_in; (void)out_size; (void)ws_size;
  const float* hs  = (const float*)d_in[0];
  const int*   pid = (const int*)d_in[1];
  const float* Wq  = (const float*)d_in[2];
  const float* Wk  = (const float*)d_in[3];
  const float* Wv  = (const float*)d_in[4];
  const float* Wo  = (const float*)d_in[5];
  float* out = (float*)d_out;

  char* ws = (char*)d_ws;
  size_t off = 0;
  auto walloc = [&](size_t bytes) -> unsigned short* {
    void* p = ws + off;
    off += (bytes + 255) & ~(size_t)255;
    return (unsigned short*)p;
  };
  unsigned short* hbf = walloc((size_t)S_LEN * HID * 2);
  unsigned short* wqt = walloc((size_t)HID * HID * 2);    // [N][K] transposed
  unsigned short* wkt = walloc((size_t)HID * KVDIM * 2);
  unsigned short* wvt = walloc((size_t)HID * KVDIM * 2);
  unsigned short* wot = walloc((size_t)HID * HID * 2);
  unsigned short* Qb  = walloc((size_t)S_LEN * HID * 2);
  unsigned short* Kb  = walloc((size_t)S_LEN * KVDIM * 2);
  unsigned short* VtG = walloc((size_t)S_LEN * KVDIM * 2); // V^T [KVDIM][S]
  unsigned short* Ab  = walloc((size_t)S_LEN * HID * 2);

  {  // activations: straight convert
    int n8 = (int)(((size_t)S_LEN * HID) / 8);
    cvt_f32_bf16<<<(n8 + 255) / 256, 256, 0, stream>>>(hs, hbf, n8);
  }
  auto cvtT = [&](const float* src, unsigned short* dst, int K, int N) {
    int n8 = (K * N) / 8;
    cvt_f32_bf16_T<<<(n8 + 255) / 256, 256, 0, stream>>>(src, dst, K, N);
  };
  cvtT(Wq, wqt, HID, HID);
  cvtT(Wk, wkt, HID, KVDIM);
  cvtT(Wv, wvt, HID, KVDIM);
  cvtT(Wo, wot, HID, HID);

  // QKV projections (bf16 out; V written transposed for flash)
  gemm_bf16<0><<<dim3(HID / 128, S_LEN / 128), 256, 0, stream>>>(
      hbf, wqt, Qb, HID, HID);
  gemm_bf16<0><<<dim3(KVDIM / 128, S_LEN / 128), 256, 0, stream>>>(
      hbf, wkt, Kb, KVDIM, HID);
  gemm_bf16<2><<<dim3(KVDIM / 128, S_LEN / 128), 256, 0, stream>>>(
      hbf, wvt, VtG, KVDIM, HID);

  // RoPE in place on Q and K
  rope_kernel<<<(S_LEN * NH * 64) / 256, 256, 0, stream>>>(Qb, pid, NH);
  rope_kernel<<<(S_LEN * NKV * 64) / 256, 256, 0, stream>>>(Kb, pid, NKV);

  // causal flash attention -> Ab (bf16 [S][HID])
  flash_attn<<<dim3(S_LEN / 64, NH), 128, 0, stream>>>(Qb, Kb, VtG, Ab);

  // output projection (f32 out)
  gemm_bf16<1><<<dim3(HID / 128, S_LEN / 128), 256, 0, stream>>>(
      Ab, wot, (void*)out, HID, HID);
}
